// PixelDecoderTransformer_28329604284443
// MI455X (gfx1250) — matrix operand order repaired and checked by reading, not compiled, not run on hardware
//
#include <hip/hip_runtime.h>
#include <hip/hip_bf16.h>

typedef _Float16 f16_t;
typedef _Float16 v16h __attribute__((ext_vector_type(16)));
typedef _Float16 v8h  __attribute__((ext_vector_type(8)));
typedef float    v8f  __attribute__((ext_vector_type(8)));

#define DEVINL static __device__ __forceinline__

// ---------------- problem constants ----------------
constexpr int  TOK    = 64;      // K*K tokens per patch
constexpr int  CH     = 256;     // C == HID
constexpr int  NCLS   = 19;
constexpr int  NPATCH = 2048;    // B * L
// LDS strides
constexpr int  LDA  = 264;       // f16 activation stride (16B aligned rows)
constexpr int  LDP  = 72;        // f16 per-head stride
constexpr int  LDSC = 68;        // f32 score stride

// ---------------- workspace layout (offsets in halfs) ----------------
constexpr size_t OFF_CWV = 0;
constexpr size_t OFF_CWO = OFF_CWV + 256*256;
constexpr size_t OFF_CW1 = OFF_CWO + 256*256;          // Wt [1024][256]
constexpr size_t OFF_CW2 = OFF_CW1 + 1024*256;         // Wt [256][1024]
constexpr size_t OFF_MAP = OFF_CW2 + 1024*256;
constexpr size_t OFF_SQ  = OFF_MAP + 256*256;          // 2 layers
constexpr size_t OFF_SK  = OFF_SQ  + 2*256*256;
constexpr size_t OFF_SV  = OFF_SK  + 2*256*256;
constexpr size_t OFF_SO  = OFF_SV  + 2*256*256;
constexpr size_t OFF_SW1 = OFF_SO  + 2*256*256;        // 2 x [1024][256]
constexpr size_t OFF_SW2 = OFF_SW1 + 2*1024*256;       // 2 x [256][1024]
constexpr size_t OFF_CLS = OFF_SW2 + 2*1024*256;       // [32][256] zero-padded
constexpr size_t OFF_PE  = OFF_CLS + 32*256;           // pe2d f16 [64*64][64]
constexpr size_t OFF_AA  = OFF_PE  + 64*64*64;         // attn_add f16 [2048][256]

// ---------------- WMMA helpers ----------------
DEVINL v8f zero8() {
  v8f r;
#pragma unroll
  for (int i = 0; i < 8; i++) r[i] = 0.f;
  return r;
}

DEVINL v8f wmma_f16(v16h a, v16h b, v8f c) {
  return __builtin_amdgcn_wmma_f32_16x16x32_f16(false, a, false, b, (short)0, c,
                                                false, false);
}

// A fragment: 16x32 f16 tile. lane L holds row m0+(L&15); K halves:
// lanes 0-15 -> K {0..7, 16..23}, lanes 16-31 -> +8.  Two 16B loads.
DEVINL v16h load_a(const f16_t* A, int lda, int m0, int k0, int lane) {
  const f16_t* p = A + (size_t)(m0 + (lane & 15)) * lda + k0 + ((lane & 16) ? 8 : 0);
  v8h lo = *(const v8h*)p;
  v8h hi = *(const v8h*)(p + 16);
  return __builtin_shufflevector(lo, hi, 0,1,2,3,4,5,6,7,8,9,10,11,12,13,14,15);
}

// B fragment from K-contiguous (transposed) weight Bt[n][k]: lane L holds
// col n0+(L&15); lanes 0-15 -> K {0..7, 16..23}, lanes 16-31 -> +8.
DEVINL v16h load_b(const f16_t* Bt, int ldb, int n0, int k0, int lane) {
  const f16_t* p = Bt + (size_t)(n0 + (lane & 15)) * ldb + k0 + ((lane & 16) ? 8 : 0);
  v8h lo = *(const v8h*)p;
  v8h hi = *(const v8h*)(p + 16);
  return __builtin_shufflevector(lo, hi, 0,1,2,3,4,5,6,7,8,9,10,11,12,13,14,15);
}

// C/D layout: VGPR r -> row m0 + r + (lane>=16 ? 8 : 0), col n0 + (lane&15)
DEVINL void store_f32(float* C, int ldc, int m0, int n0, v8f acc,
                      const float* bias, int lane) {
  int col = n0 + (lane & 15);
  float bv = bias ? bias[col] : 0.f;
  int rb = m0 + ((lane & 16) ? 8 : 0);
#pragma unroll
  for (int r = 0; r < 8; r++) C[(rb + r) * ldc + col] = acc[r] + bv;
}

DEVINL void store_f16x(f16_t* C, int ldc, int m0, int n0, v8f acc,
                       const float* bias, bool relu, int lane) {
  int col = n0 + (lane & 15);
  float bv = bias ? bias[col] : 0.f;
  int rb = m0 + ((lane & 16) ? 8 : 0);
#pragma unroll
  for (int r = 0; r < 8; r++) {
    float v = acc[r] + bv;
    if (relu) v = fmaxf(v, 0.f);
    C[(rb + r) * ldc + col] = (f16_t)v;
  }
}

template <int NPW>
DEVINL void gemm_acc(const f16_t* A, int lda, const f16_t* Bt, int ldb, int kdim,
                     int m0, int n0, v8f* acc, int lane) {
  for (int ks = 0; ks < kdim; ks += 32) {
    v16h a = load_a(A, lda, m0, ks, lane);
#pragma unroll
    for (int j = 0; j < NPW; j++) {
      v16h b = load_b(Bt, ldb, n0 + j * 16, ks, lane);
      acc[j] = wmma_f16(a, b, acc[j]);
    }
  }
}

// Small M=64,N=64 GEMM: 16 tiles over 8 waves, 2 tiles each.
// MODE 0: f16 row-major out (+colOff), MODE 1: f16 transposed out, MODE 2: f32 out
template <int MODE>
DEVINL void gemm_n64(const f16_t* A, int lda, const f16_t* Bt, int ldb, int kdim,
                     f16_t* Ch, float* Cf, int ldc, int colOff,
                     const float* bias, int wv, int lane) {
#pragma unroll
  for (int u = 0; u < 2; u++) {
    int t = wv * 2 + u;
    int m0 = (t & 3) * 16;
    int n0 = (t >> 2) * 16;
    v8f acc = zero8();
    for (int ks = 0; ks < kdim; ks += 32) {
      v16h a = load_a(A, lda, m0, ks, lane);
      v16h b = load_b(Bt, ldb, n0, ks, lane);
      acc = wmma_f16(a, b, acc);
    }
    int col = n0 + (lane & 15);
    float bv = bias ? bias[col] : 0.f;
    int rb = m0 + ((lane & 16) ? 8 : 0);
    if (MODE == 0) {
#pragma unroll
      for (int r = 0; r < 8; r++)
        Ch[(rb + r) * ldc + colOff + col] = (f16_t)(acc[r] + bv);
    } else if (MODE == 1) {
#pragma unroll
      for (int r = 0; r < 8; r++)
        Ch[(colOff + col) * ldc + rb + r] = (f16_t)(acc[r] + bv);
    } else {
#pragma unroll
      for (int r = 0; r < 8; r++) Cf[(rb + r) * ldc + col] = acc[r];
    }
  }
}

// LayerNorm over 256-wide rows; 4 threads per row.
DEVINL void ln_rows(const float* Yf, f16_t* Xh, int addX, const float* g,
                    const float* b, float* redS, float* redQ, float* mrow,
                    float* rsrow, int tid) {
  int row = tid >> 2, part = tid & 3;
  float s = 0.f, q = 0.f;
  for (int i = 0; i < 64; i++) {
    int c = part * 64 + i;
    float v = Yf[row * 256 + c];
    if (addX) v += (float)Xh[row * LDA + c];
    s += v; q += v * v;
  }
  redS[tid] = s; redQ[tid] = q;
  __syncthreads();
  if (part == 0) {
    float S = redS[row*4] + redS[row*4+1] + redS[row*4+2] + redS[row*4+3];
    float Q = redQ[row*4] + redQ[row*4+1] + redQ[row*4+2] + redQ[row*4+3];
    float m = S * (1.f / 256.f);
    float var = Q * (1.f / 256.f) - m * m;
    mrow[row] = m;
    rsrow[row] = rsqrtf(var + 1e-5f);
  }
  __syncthreads();
  float m = mrow[row], rs = rsrow[row];
  for (int i = 0; i < 64; i++) {
    int c = part * 64 + i;
    float v = Yf[row * 256 + c];
    if (addX) v += (float)Xh[row * LDA + c];
    Xh[row * LDA + c] = (f16_t)((v - m) * rs * g[c] + b[c]);
  }
  __syncthreads();
}

// Chunked FFN: hidden 1024 processed in 4 chunks of 256 (hidden chunk in LDS,
// output accumulators held in registers across chunks), then residual + LN.
DEVINL void ffn_block(f16_t* Xh, f16_t* Bufh, float* Yf, const f16_t* W1t,
                      const float* b1, const f16_t* W2t, const float* b2,
                      const float* g, const float* bb, float* redS, float* redQ,
                      float* mrow, float* rsrow, int wv, int lane, int tid) {
  int m0 = (wv & 3) * 16, nb = (wv >> 2) * 128;
  v8f acc[8];
#pragma unroll
  for (int j = 0; j < 8; j++) acc[j] = zero8();
  for (int c = 0; c < 4; c++) {
    __syncthreads();
    v8f a1[8];
#pragma unroll
    for (int j = 0; j < 8; j++) a1[j] = zero8();
    gemm_acc<8>(Xh, LDA, W1t + (size_t)c * 256 * 256, 256, 256, m0, nb, a1, lane);
#pragma unroll
    for (int j = 0; j < 8; j++)
      store_f16x(Bufh, LDA, m0, nb + j * 16, a1[j], b1 + c * 256, true, lane);
    __syncthreads();
    gemm_acc<8>(Bufh, LDA, W2t + c * 256, 1024, 256, m0, nb, acc, lane);
  }
  __syncthreads();
#pragma unroll
  for (int j = 0; j < 8; j++)
    store_f32(Yf, 256, m0, nb + j * 16, acc[j], b2, lane);
  __syncthreads();
  ln_rows(Yf, Xh, 1, g, bb, redS, redQ, mrow, rsrow, tid);
}

// ---------------- main fused per-patch kernel ----------------
struct MainParams {
  const float* x;
  const float* pixel_W;
  const float* pixel_b;
  const float* c_ln1_g; const float* c_ln1_b;
  const float* c_b1;    const float* c_b2;
  const float* c_ln2_g; const float* c_ln2_b;
  const float* map_b;
  const float* s_bq; const float* s_bk; const float* s_bv; const float* s_bo;
  const float* s_ln1_g; const float* s_ln1_b;
  const float* s_b1; const float* s_b2;
  const float* s_ln2_g; const float* s_ln2_b;
  const float* cls_b;
  const f16_t* ws;
  float* out;
};

DEVINL void self_block(const MainParams& P, int layer, f16_t* Xh, f16_t* Bufh,
                       unsigned char* U, float* redS, float* redQ, float* mrow,
                       float* rsrow, int wv, int lane, int tid) {
  f16_t* qh = (f16_t*)U;                 // [64][72] f16
  f16_t* kh = (f16_t*)(U + 9216);        // [64][72] f16
  f16_t* vT = (f16_t*)(U + 18432);       // [64(d)][72] f16 (transposed)
  float* Sf = (float*)(U + 27648);       // [64][68] f32
  f16_t* Pp = (f16_t*)(U + 45056);       // [64][72] f16 probs
  float* Yf = (float*)U;                 // [64][256] f32 (aliases qh..Pp)

  const f16_t* Wq = P.ws + OFF_SQ + (size_t)layer * 65536;
  const f16_t* Wk = P.ws + OFF_SK + (size_t)layer * 65536;
  const f16_t* Wv = P.ws + OFF_SV + (size_t)layer * 65536;
  const f16_t* Wo = P.ws + OFF_SO + (size_t)layer * 65536;
  const f16_t* W1 = P.ws + OFF_SW1 + (size_t)layer * 262144;
  const f16_t* W2 = P.ws + OFF_SW2 + (size_t)layer * 262144;
  const f16_t* pe = P.ws + OFF_PE;
  const float* bq = P.s_bq + layer * 256;
  const float* bk = P.s_bk + layer * 256;
  const float* bv = P.s_bv + layer * 256;
  const float* bo = P.s_bo + layer * 256;
  const float* b1 = P.s_b1 + layer * 1024;
  const float* b2 = P.s_b2 + layer * 256;
  const float* g1 = P.s_ln1_g + layer * 256;
  const float* l1 = P.s_ln1_b + layer * 256;
  const float* g2 = P.s_ln2_g + layer * 256;
  const float* l2 = P.s_ln2_b + layer * 256;

  for (int h = 0; h < 4; h++) {
    __syncthreads();
    gemm_n64<0>(Xh, LDA, Wq + (size_t)h * 64 * 256, 256, 256, qh, nullptr, LDP, 0, bq + h * 64, wv, lane);
    gemm_n64<0>(Xh, LDA, Wk + (size_t)h * 64 * 256, 256, 256, kh, nullptr, LDP, 0, bk + h * 64, wv, lane);
    gemm_n64<1>(Xh, LDA, Wv + (size_t)h * 64 * 256, 256, 256, vT, nullptr, LDP, 0, bv + h * 64, wv, lane);
    __syncthreads();
    // scores = q @ k^T  (K-dim = head dim 64; kh rows are K-contiguous)
    gemm_n64<2>(qh, LDP, kh, LDP, 64, nullptr, Sf, LDSC, 0, nullptr, wv, lane);
    __syncthreads();
    // relative-position bias: S[q,k] += dot(q[q,:], pe2d[q,k,:]); then scale
    {
      int q = tid >> 2, kb = (tid & 3) * 16;
      const f16_t* qrow = qh + q * LDP;
      for (int j = 0; j < 16; j++) {
        int k = kb + j;
        const f16_t* pp = pe + ((size_t)(q * 64 + k) << 6);
        float dot = 0.f;
#pragma unroll
        for (int d = 0; d < 64; d++) dot += (float)qrow[d] * (float)pp[d];
        Sf[q * LDSC + k] = (Sf[q * LDSC + k] + dot) * 0.125f;
      }
    }
    __syncthreads();
    if (tid < 64) {
      float mx = -3.0e38f;
      for (int k = 0; k < 64; k++) mx = fmaxf(mx, Sf[tid * LDSC + k]);
      float sm = 0.f;
      for (int k = 0; k < 64; k++) sm += __expf(Sf[tid * LDSC + k] - mx);
      float inv = 1.f / sm;
      for (int k = 0; k < 64; k++)
        Pp[tid * LDP + k] = (f16_t)(__expf(Sf[tid * LDSC + k] - mx) * inv);
    }
    __syncthreads();
    // attnout_h = P @ v  -> att buffer columns h*64..h*64+63
    gemm_n64<0>(Pp, LDP, vT, LDP, 64, Bufh, nullptr, LDA, h * 64, nullptr, wv, lane);
  }
  __syncthreads();
  {
    int m0 = (wv & 3) * 16, nb = (wv >> 2) * 128;
    v8f acc[8];
#pragma unroll
    for (int j = 0; j < 8; j++) acc[j] = zero8();
    gemm_acc<8>(Bufh, LDA, Wo, 256, 256, m0, nb, acc, lane);
#pragma unroll
    for (int j = 0; j < 8; j++)
      store_f32(Yf, 256, m0, nb + j * 16, acc[j], bo, lane);
  }
  __syncthreads();
  ln_rows(Yf, Xh, 1, g1, l1, redS, redQ, mrow, rsrow, tid);
  ffn_block(Xh, Bufh, Yf, W1, b1, W2, b2, g2, l2, redS, redQ, mrow, rsrow, wv, lane, tid);
}

__global__ void __launch_bounds__(256, 1) k_main(MainParams P) {
  __shared__ __align__(16) f16_t Xh[64 * LDA];
  __shared__ __align__(16) f16_t Bufh[64 * LDA];
  __shared__ __align__(16) unsigned char U[65536];
  __shared__ float redS[256], redQ[256], mrow[64], rsrow[64];

  const int tid = threadIdx.x, lane = tid & 31, wv = tid >> 5;
  const int n = blockIdx.x;
  const int b = n >> 10, l = n & 1023, py = l >> 5, px = l & 31;
  float* Yf = (float*)U;

  // ---- phase 0: pixel projection + precomputed cross-attn add, LN -> Xh ----
  {
    int row = tid >> 2, part = tid & 3;
    int ky = row >> 3, kx = row & 7;
    int hh = py * 8 + ky, ww = px * 8 + kx;
    float x0 = P.x[((size_t)(b * 3 + 0) * 256 + hh) * 256 + ww];
    float x1 = P.x[((size_t)(b * 3 + 1) * 256 + hh) * 256 + ww];
    float x2 = P.x[((size_t)(b * 3 + 2) * 256 + hh) * 256 + ww];
    const f16_t* aadd = P.ws + OFF_AA + (size_t)n * 256;
    for (int i = 0; i < 64; i++) {
      int c = part * 64 + i;
      float v = x0 * P.pixel_W[c] + x1 * P.pixel_W[256 + c] +
                x2 * P.pixel_W[512 + c] + P.pixel_b[c] + (float)aadd[c];
      Yf[row * 256 + c] = v;
    }
    __syncthreads();
    ln_rows(Yf, Xh, 0, P.c_ln1_g, P.c_ln1_b, redS, redQ, mrow, rsrow, tid);
  }
  // ---- phase 1: cross FFN ----
  ffn_block(Xh, Bufh, Yf, P.ws + OFF_CW1, P.c_b1, P.ws + OFF_CW2, P.c_b2,
            P.c_ln2_g, P.c_ln2_b, redS, redQ, mrow, rsrow, wv, lane, tid);
  // ---- phase 2: map projection (no residual, no LN) ----
  {
    int m0 = (wv & 3) * 16, nb = (wv >> 2) * 128;
    v8f acc[8];
#pragma unroll
    for (int j = 0; j < 8; j++) acc[j] = zero8();
    gemm_acc<8>(Xh, LDA, P.ws + OFF_MAP, 256, 256, m0, nb, acc, lane);
#pragma unroll
    for (int j = 0; j < 8; j++)
      store_f32(Yf, 256, m0, nb + j * 16, acc[j], P.map_b, lane);
    __syncthreads();
    for (int e = tid; e < 64 * 256; e += 256)
      Xh[(e >> 8) * LDA + (e & 255)] = (f16_t)Yf[e];
    __syncthreads();
  }
  // ---- phase 3: self-attention transformer layers ----
  self_block(P, 0, Xh, Bufh, U, redS, redQ, mrow, rsrow, wv, lane, tid);
  self_block(P, 1, Xh, Bufh, U, redS, redQ, mrow, rsrow, wv, lane, tid);
  // ---- phase 4: classifier + fold ----
  {
    int m0 = (wv & 3) * 16, n0 = (wv >> 2) * 16;   // 8 tiles over 8 waves
    v8f acc = zero8();
    gemm_acc<1>(Xh, LDA, P.ws + OFF_CLS, 256, 256, m0, n0, &acc, lane);
    store_f32(Yf, 32, m0, n0, acc, nullptr, lane);
    __syncthreads();
    for (int e = tid; e < 64 * NCLS; e += 256) {
      int t = e / NCLS, nc = e - t * NCLS;
      int ky = t >> 3, kx = t & 7;
      P.out[(((size_t)b * NCLS + nc) * 256 + py * 8 + ky) * 256 + px * 8 + kx] =
          Yf[t * 32 + nc] + P.cls_b[nc];
    }
  }
}

// ---------------- setup kernels ----------------
__global__ void k_convtr(const float* __restrict__ src, f16_t* __restrict__ dst,
                         int Kd, int N) {
  int idx = blockIdx.x * 256 + threadIdx.x;
  if (idx >= Kd * N) return;
  int k = idx / N, nn = idx - k * N;
  dst[(size_t)nn * Kd + k] = (f16_t)src[idx];
}

__global__ void k_zero16(f16_t* dst, int n) {
  int i = blockIdx.x * 256 + threadIdx.x;
  if (i < n) dst[i] = (f16_t)0.f;
}

__global__ void k_pe2d(const float* __restrict__ pex, const float* __restrict__ pey,
                       f16_t* __restrict__ dst) {
  int idx = blockIdx.x * 256 + threadIdx.x;   // 64*64*64 total
  int d = idx & 63, k = (idx >> 6) & 63, q = idx >> 12;
  int xi = q & 7, yi = q >> 3, xj = k & 7, yj = k >> 3;
  int xd = (xj - xi) + 8, yd = (yj - yi) + 8;
  dst[idx] = (f16_t)(pex[xd * 64 + d] + pey[yd * 64 + d]);
}

// attn_add[n,:] = (embed[n] @ Wv + bv) @ Wo + bo   (cross-attn with Lk==1)
__global__ void __launch_bounds__(256, 1)
k_attn_add(const float* __restrict__ embed, const f16_t* __restrict__ WvT,
           const float* __restrict__ bv, const f16_t* __restrict__ WoT,
           const float* __restrict__ bo, f16_t* __restrict__ outAdd) {
  __shared__ __align__(16) f16_t Eh[64 * LDA];
  __shared__ __align__(16) f16_t Vh[64 * LDA];
  const int tid = threadIdx.x, lane = tid & 31, wv = tid >> 5;
  const size_t n0 = (size_t)blockIdx.x * 64;
  for (int e = tid; e < 64 * 256; e += 256)
    Eh[(e >> 8) * LDA + (e & 255)] = (f16_t)embed[n0 * 256 + e];
  __syncthreads();
  int m0 = (wv & 3) * 16, nb = (wv >> 2) * 128;
  {
    v8f acc[8];
#pragma unroll
    for (int j = 0; j < 8; j++) acc[j] = zero8();
    gemm_acc<8>(Eh, LDA, WvT, 256, 256, m0, nb, acc, lane);
#pragma unroll
    for (int j = 0; j < 8; j++)
      store_f16x(Vh, LDA, m0, nb + j * 16, acc[j], bv, false, lane);
  }
  __syncthreads();
  {
    v8f acc[8];
#pragma unroll
    for (int j = 0; j < 8; j++) acc[j] = zero8();
    gemm_acc<8>(Vh, LDA, WoT, 256, 256, m0, nb, acc, lane);
#pragma unroll
    for (int j = 0; j < 8; j++) {
      int col = nb + j * 16 + (lane & 15);
      float bvv = bo[col];
      int rb = m0 + ((lane & 16) ? 8 : 0);
#pragma unroll
      for (int r = 0; r < 8; r++)
        outAdd[(n0 + rb + r) * 256 + col] = (f16_t)(acc[j][r] + bvv);
    }
  }
}

// ---------------- launch ----------------
extern "C" void kernel_launch(void* const* d_in, const int* in_sizes, int n_in,
                              void* d_out, int out_size, void* d_ws, size_t ws_size,
                              hipStream_t stream) {
  f16_t* W = (f16_t*)d_ws;
  const float* embed   = (const float*)d_in[0];
  const float* x       = (const float*)d_in[1];
  const float* pixel_W = (const float*)d_in[2];
  const float* pixel_b = (const float*)d_in[3];
  const float* pe2d_x  = (const float*)d_in[6];
  const float* pe2d_y  = (const float*)d_in[7];
  const float* c_Wv    = (const float*)d_in[12];
  const float* c_bv    = (const float*)d_in[13];
  const float* c_Wo    = (const float*)d_in[14];
  const float* c_bo    = (const float*)d_in[15];
  const float* c_ln1_g = (const float*)d_in[16];
  const float* c_ln1_b = (const float*)d_in[17];
  const float* c_W1    = (const float*)d_in[18];
  const float* c_b1    = (const float*)d_in[19];
  const float* c_W2    = (const float*)d_in[20];
  const float* c_b2    = (const float*)d_in[21];
  const float* c_ln2_g = (const float*)d_in[22];
  const float* c_ln2_b = (const float*)d_in[23];
  const float* map_W   = (const float*)d_in[24];
  const float* map_b   = (const float*)d_in[25];
  const float* s_Wq    = (const float*)d_in[26];
  const float* s_bq    = (const float*)d_in[27];
  const float* s_Wk    = (const float*)d_in[28];
  const float* s_bk    = (const float*)d_in[29];
  const float* s_Wv    = (const float*)d_in[30];
  const float* s_bv    = (const float*)d_in[31];
  const float* s_Wo    = (const float*)d_in[32];
  const float* s_bo    = (const float*)d_in[33];
  const float* s_ln1_g = (const float*)d_in[34];
  const float* s_ln1_b = (const float*)d_in[35];
  const float* s_W1    = (const float*)d_in[36];
  const float* s_b1    = (const float*)d_in[37];
  const float* s_W2    = (const float*)d_in[38];
  const float* s_b2    = (const float*)d_in[39];
  const float* s_ln2_g = (const float*)d_in[40];
  const float* s_ln2_b = (const float*)d_in[41];
  const float* cls_W   = (const float*)d_in[42];
  const float* cls_b   = (const float*)d_in[43];

  auto ct = [&](const float* s, f16_t* d, int Kd, int N) {
    int tot = Kd * N;
    k_convtr<<<(tot + 255) / 256, 256, 0, stream>>>(s, d, Kd, N);
  };
  ct(c_Wv, W + OFF_CWV, 256, 256);
  ct(c_Wo, W + OFF_CWO, 256, 256);
  ct(c_W1, W + OFF_CW1, 256, 1024);
  ct(c_W2, W + OFF_CW2, 1024, 256);
  ct(map_W, W + OFF_MAP, 256, 256);
  for (int lyr = 0; lyr < 2; lyr++) {
    ct(s_Wq + lyr * 65536, W + OFF_SQ + (size_t)lyr * 65536, 256, 256);
    ct(s_Wk + lyr * 65536, W + OFF_SK + (size_t)lyr * 65536, 256, 256);
    ct(s_Wv + lyr * 65536, W + OFF_SV + (size_t)lyr * 65536, 256, 256);
    ct(s_Wo + lyr * 65536, W + OFF_SO + (size_t)lyr * 65536, 256, 256);
    ct(s_W1 + lyr * 262144, W + OFF_SW1 + (size_t)lyr * 262144, 256, 1024);
    ct(s_W2 + lyr * 262144, W + OFF_SW2 + (size_t)lyr * 262144, 1024, 256);
  }
  k_zero16<<<32, 256, 0, stream>>>(W + OFF_CLS, 32 * 256);
  ct(cls_W, W + OFF_CLS, 256, NCLS);
  k_pe2d<<<(64 * 64 * 64) / 256, 256, 0, stream>>>(pe2d_x, pe2d_y, W + OFF_PE);
  k_attn_add<<<32, 256, 0, stream>>>(embed, W + OFF_CWV, c_bv, W + OFF_CWO, c_bo,
                                     W + OFF_AA);

  MainParams P;
  P.x = x; P.pixel_W = pixel_W; P.pixel_b = pixel_b;
  P.c_ln1_g = c_ln1_g; P.c_ln1_b = c_ln1_b;
  P.c_b1 = c_b1; P.c_b2 = c_b2;
  P.c_ln2_g = c_ln2_g; P.c_ln2_b = c_ln2_b;
  P.map_b = map_b;
  P.s_bq = s_bq; P.s_bk = s_bk; P.s_bv = s_bv; P.s_bo = s_bo;
  P.s_ln1_g = s_ln1_g; P.s_ln1_b = s_ln1_b;
  P.s_b1 = s_b1; P.s_b2 = s_b2;
  P.s_ln2_g = s_ln2_g; P.s_ln2_b = s_ln2_b;
  P.cls_b = cls_b;
  P.ws = W;
  P.out = (float*)d_out;
  k_main<<<NPATCH, 256, 0, stream>>>(P);

  (void)in_sizes; (void)n_in; (void)out_size; (void)ws_size;
}
